// LlamaAttention_627065225528
// MI455X (gfx1250) — matrix operand order repaired and checked by reading, not compiled, not run on hardware
//
#include <hip/hip_runtime.h>
#include <hip/hip_bf16.h>

// ---------------------------------------------------------------------------
// LlamaAttention forward for MI455X (gfx1250), bf16 WMMA path.
//   B=2, S=2048, H=4096, 32 heads x 128 dim, 8 KV heads (GQA group 4).
// Pipeline: f32->bf16 convert, QKV GEMMs (bf16 WMMA, f32 acc, TDM-fed double-
// buffered LDS), flash attention (WMMA QK^T + online softmax + WMMA PV),
// O-projection GEMM (f32 out).
// ---------------------------------------------------------------------------

typedef __bf16 bf16_t;
typedef __attribute__((ext_vector_type(16))) __bf16 v16bf;
typedef __attribute__((ext_vector_type(8)))  float  v8f;
typedef __attribute__((ext_vector_type(4)))  unsigned int v4u;
typedef __attribute__((ext_vector_type(8)))  int  v8i;
typedef __attribute__((ext_vector_type(4)))  int  v4i;

// Tensor Data Mover availability (6-arg upstream clang form on this toolchain).
#if defined(__AMDGCN__) && __has_builtin(__builtin_amdgcn_tensor_load_to_lds) && \
    __has_builtin(__builtin_amdgcn_s_wait_tensorcnt)
#define USE_TDM 1
#else
#define USE_TDM 0
#endif

union BFrag {
    unsigned int u[8];
    v16bf        v;
};
static_assert(sizeof(BFrag) == 32, "frag size");

__device__ inline v8f zero8() {
    v8f r = {0.f, 0.f, 0.f, 0.f, 0.f, 0.f, 0.f, 0.f};
    return r;
}

__device__ inline v8f wmma_bf16(const BFrag& a, const BFrag& b, v8f c) {
    // (neg_a, A, neg_b, B, c_mod, C, reuse_a, reuse_b)
    return __builtin_amdgcn_wmma_f32_16x16x32_bf16(false, a.v, false, b.v,
                                                   (short)0, c, false, false);
}

// A fragment: 16x32 bf16 (M x K). Lane l holds row m=l&15; K split:
// VGPR 0..3 -> K = 8*(l>>4) + {2v,2v+1}; VGPR 4..7 -> K = 16 + same.
__device__ inline void load_a_frag(const bf16_t* __restrict__ base, int ld,
                                   int row0, int k0, BFrag& f) {
    const int lane = threadIdx.x & 31;
    const unsigned int* p =
        (const unsigned int*)(base + (size_t)(row0 + (lane & 15)) * ld + k0);
    const int kd = (lane >> 4) * 4;
#pragma unroll
    for (int v = 0; v < 4; ++v) {
        f.u[v]     = p[kd + v];
        f.u[4 + v] = p[8 + kd + v];
    }
}

// B fragment: 32x16 bf16 (K x N). Lane l holds col n=l&15; K split:
// VGPR 0..3 -> K = 8*(l>>4) + {2v,2v+1}; VGPR 4..7 -> K = 16 + same.
// Source rows are the N dimension: base[n][k], row-major, leading dim `ld`.
__device__ inline void load_b_frag(const bf16_t* __restrict__ base, int ld,
                                   int n0, int k0, BFrag& f) {
    const int lane = threadIdx.x & 31;
    const unsigned int* p =
        (const unsigned int*)(base + (size_t)(n0 + (lane & 15)) * ld + k0);
    const int kd = (lane >> 4) * 4;
#pragma unroll
    for (int v = 0; v < 4; ++v) {
        f.u[v]     = p[kd + v];
        f.u[4 + v] = p[8 + kd + v];
    }
}

#if USE_TDM
// Issue one TDM 2D tile load: tileX contiguous bf16 elements per row,
// tileY rows, row stride strideElems (bf16 elements), into LDS at lds_addr.
// D# packing per CDNA5 ISA.
__device__ inline void tdm_load_2d(unsigned int lds_addr,
                                   const bf16_t* gptr, unsigned int tileX,
                                   unsigned int tileY,
                                   unsigned long long strideElems) {
    const unsigned long long ga = (unsigned long long)(uintptr_t)gptr;
    v4u g0;
    g0[0] = 1u;                                   // count=1, user descriptor
    g0[1] = lds_addr;                             // LDS byte address
    g0[2] = (unsigned int)ga;                     // global_addr[31:0]
    g0[3] = (unsigned int)((ga >> 32) & 0x01FFFFFFu) | 0x80000000u;  // type=2
    v8i g1;
    g1[0] = (int)(1u << 16);                      // data_size=1 (2 bytes)
    g1[1] = (int)((tileX & 0xFFFFu) << 16);       // tensor_dim0[15:0]
    g1[2] = (int)((tileX >> 16) | ((tileY & 0xFFFFu) << 16));  // dim0 hi|dim1 lo
    g1[3] = (int)((tileY >> 16) | (tileX << 16)); // dim1 hi | tile_dim0
    g1[4] = (int)(tileY & 0xFFFFu);               // tile_dim1 (tile_dim2=0)
    g1[5] = (int)(unsigned int)(strideElems & 0xFFFFFFFFull);  // dim0_stride lo
    g1[6] = (int)(unsigned int)((strideElems >> 32) & 0xFFFFull);
    g1[7] = 0;
    v4i z4 = {0, 0, 0, 0};
    v8i z8 = {0, 0, 0, 0, 0, 0, 0, 0};
    // 6-arg form: (g0, g1, g2, g3, extra, cpol)
    __builtin_amdgcn_tensor_load_to_lds(g0, g1, z4, z4, z8, 0);
}
#endif

// ---------------------------------------------------------------------------
// f32 -> bf16 conversion
// ---------------------------------------------------------------------------
__global__ void f32_to_bf16_kernel(const float* __restrict__ in,
                                   bf16_t* __restrict__ out, size_t n) {
    size_t stride = (size_t)gridDim.x * blockDim.x;
    for (size_t i = (size_t)blockIdx.x * blockDim.x + threadIdx.x; i < n;
         i += stride) {
        out[i] = (bf16_t)in[i];
    }
}

// ---------------------------------------------------------------------------
// GEMM: C[M][N] = A[M][K] * B[N][K]^T  (A,B bf16 row-major, acc f32)
// Block tile 128x128, K-step 32, 8 waves: 2(M) x 4(N). Double-buffered LDS,
// fed by TDM (one wave issues tensor_load_to_lds) or register staging.
// ---------------------------------------------------------------------------
template <bool OUT_BF16>
__global__ __launch_bounds__(256) void gemm_bt_kernel(
    const bf16_t* __restrict__ A, const bf16_t* __restrict__ B,
    void* __restrict__ Cout, int M, int N, int K) {
    __shared__ bf16_t As[2][128][32];
    __shared__ bf16_t Bs[2][128][32];

    const int tid  = threadIdx.x;
    const int wave = tid >> 5;
    const int lane = tid & 31;
    const int wm   = wave >> 2;  // 0..1
    const int wn   = wave & 3;   // 0..3
    const int mBase = blockIdx.y * 128;
    const int nBase = blockIdx.x * 128;
    const int nk    = K >> 5;

    v8f acc[4][2];
#pragma unroll
    for (int mt = 0; mt < 4; ++mt)
#pragma unroll
        for (int nt = 0; nt < 2; ++nt) acc[mt][nt] = zero8();

    int buf = 0;

#if USE_TDM
    if (wave == 0) {
        tdm_load_2d((unsigned int)(uintptr_t)&As[0][0][0],
                    A + (size_t)mBase * K, 32u, 128u, (unsigned long long)K);
        tdm_load_2d((unsigned int)(uintptr_t)&Bs[0][0][0],
                    B + (size_t)nBase * K, 32u, 128u, (unsigned long long)K);
    }
    for (int kt = 0; kt < nk; ++kt) {
        if (wave == 0) __builtin_amdgcn_s_wait_tensorcnt(0);
        __syncthreads();  // LDS[buf] fully written; prior reads of buf^1 done
        if (wave == 0 && kt + 1 < nk) {
            const int k1 = (kt + 1) << 5;
            tdm_load_2d((unsigned int)(uintptr_t)&As[buf ^ 1][0][0],
                        A + (size_t)mBase * K + k1, 32u, 128u,
                        (unsigned long long)K);
            tdm_load_2d((unsigned int)(uintptr_t)&Bs[buf ^ 1][0][0],
                        B + (size_t)nBase * K + k1, 32u, 128u,
                        (unsigned long long)K);
        }
        BFrag af[4], bfr[2];
#pragma unroll
        for (int mt = 0; mt < 4; ++mt)
            load_a_frag(&As[buf][0][0], 32, wm * 64 + mt * 16, 0, af[mt]);
#pragma unroll
        for (int nt = 0; nt < 2; ++nt)
            load_b_frag(&Bs[buf][0][0], 32, wn * 32 + nt * 16, 0, bfr[nt]);
#pragma unroll
        for (int mt = 0; mt < 4; ++mt)
#pragma unroll
            for (int nt = 0; nt < 2; ++nt)
                acc[mt][nt] = wmma_bf16(af[mt], bfr[nt], acc[mt][nt]);
        buf ^= 1;
    }
#else
    // Register-staged double buffer: prefetch tile kt+1 while computing kt.
    const int row0 = tid >> 2;             // pass 0 row
    const int col0 = (tid & 3) * 8;
    uint4 ra[2], rb[2];
    auto stage = [&](int k0) {
#pragma unroll
        for (int pass = 0; pass < 2; ++pass) {
            const int r = row0 + pass * 64;
            ra[pass] = *(const uint4*)(A + (size_t)(mBase + r) * K + k0 + col0);
            rb[pass] = *(const uint4*)(B + (size_t)(nBase + r) * K + k0 + col0);
        }
    };
    auto commit = [&](int b2) {
#pragma unroll
        for (int pass = 0; pass < 2; ++pass) {
            const int r = row0 + pass * 64;
            *(uint4*)&As[b2][r][col0] = ra[pass];
            *(uint4*)&Bs[b2][r][col0] = rb[pass];
        }
    };
    stage(0);
    commit(0);
    for (int kt = 0; kt < nk; ++kt) {
        __syncthreads();  // LDS[buf] ready; reads of LDS[buf^1] finished
        if (kt + 1 < nk) stage((kt + 1) << 5);
        BFrag af[4], bfr[2];
#pragma unroll
        for (int mt = 0; mt < 4; ++mt)
            load_a_frag(&As[buf][0][0], 32, wm * 64 + mt * 16, 0, af[mt]);
#pragma unroll
        for (int nt = 0; nt < 2; ++nt)
            load_b_frag(&Bs[buf][0][0], 32, wn * 32 + nt * 16, 0, bfr[nt]);
#pragma unroll
        for (int mt = 0; mt < 4; ++mt)
#pragma unroll
            for (int nt = 0; nt < 2; ++nt)
                acc[mt][nt] = wmma_bf16(af[mt], bfr[nt], acc[mt][nt]);
        if (kt + 1 < nk) commit(buf ^ 1);
        buf ^= 1;
    }
#endif

    // Epilogue. C layout: VGPR i -> row (i + 8*(lane>>4)), col = lane&15.
    const int half = (lane >> 4) * 8;
    const int nl   = lane & 15;
#pragma unroll
    for (int mt = 0; mt < 4; ++mt) {
#pragma unroll
        for (int nt = 0; nt < 2; ++nt) {
#pragma unroll
            for (int i = 0; i < 8; ++i) {
                const int r = mBase + wm * 64 + mt * 16 + i + half;
                const int c = nBase + wn * 32 + nt * 16 + nl;
                if (r < M && c < N) {
                    if (OUT_BF16)
                        ((bf16_t*)Cout)[(size_t)r * N + c] =
                            (bf16_t)acc[mt][nt][i];
                    else
                        ((float*)Cout)[(size_t)r * N + c] = acc[mt][nt][i];
                }
            }
        }
    }
}

// ---------------------------------------------------------------------------
// Flash attention (no mask). Grid: (S/128, 32 heads, B). Block: 256 = 8 waves.
// Wave w owns q rows [blk*128 + w*16, +16). KV streamed in 32-key chunks with
// register prefetch of the next chunk overlapped with WMMA/softmax.
// ---------------------------------------------------------------------------
__global__ __launch_bounds__(256) void attn_kernel(
    const bf16_t* __restrict__ q, const bf16_t* __restrict__ k,
    const bf16_t* __restrict__ v, bf16_t* __restrict__ o, int S) {
    __shared__ bf16_t Ks[32][128];          // K chunk, row = key j, col = d
    __shared__ unsigned short Vts[128][32]; // V chunk transposed: [d][j]
    __shared__ bf16_t Ps[8][16][32];        // per-wave P tile

    const int tid  = threadIdx.x;
    const int wave = tid >> 5;
    const int lane = tid & 31;
    const int h    = blockIdx.y;
    const int b    = blockIdx.z;
    const int kv   = h >> 2;                // GQA: 4 q heads per kv head
    const int sRow = blockIdx.x * 128 + wave * 16;

    const int HQ  = 32 * 128;   // 4096
    const int HKV = 8 * 128;    // 1024
    const float scale = 0.08838834764831845f;  // 1/sqrt(128)

    // Q fragments for this wave's 16 rows, all 128 dims (4 K-chunks of 32).
    BFrag qf[4];
    const bf16_t* qbase = q + (size_t)(b * S + sRow) * HQ + h * 128;
#pragma unroll
    for (int dc = 0; dc < 4; ++dc) load_a_frag(qbase, HQ, 0, dc * 32, qf[dc]);

    v8f oacc[8];
#pragma unroll
    for (int nt = 0; nt < 8; ++nt) oacc[nt] = zero8();
    float mrow[8], srow[8];
#pragma unroll
    for (int i = 0; i < 8; ++i) { mrow[i] = -1e30f; srow[i] = 0.f; }

    // Chunk staging registers (each thread: one key row, 16 dims).
    const int krow = tid >> 3;           // key 0..31
    const int kcol = (tid & 7) * 16;     // d base 0..112
    uint4 kreg[2], vreg[2];
    auto stage_kv = [&](int j0) {
        const size_t grow = (size_t)(b * S + j0 + krow);
        const uint4* kg = (const uint4*)(k + grow * HKV + kv * 128 + kcol);
        kreg[0] = kg[0];
        kreg[1] = kg[1];
        const uint4* vg = (const uint4*)(v + grow * HKV + kv * 128 + kcol);
        vreg[0] = vg[0];
        vreg[1] = vg[1];
    };
    stage_kv(0);

    for (int j0 = 0; j0 < S; j0 += 32) {
        __syncthreads();  // previous iteration's Ks/Vts/Ps reads complete
        {
            *(uint4*)&Ks[krow][kcol]     = kreg[0];
            *(uint4*)&Ks[krow][kcol + 8] = kreg[1];
            unsigned int w[8];
            w[0] = vreg[0].x; w[1] = vreg[0].y; w[2] = vreg[0].z; w[3] = vreg[0].w;
            w[4] = vreg[1].x; w[5] = vreg[1].y; w[6] = vreg[1].z; w[7] = vreg[1].w;
#pragma unroll
            for (int e = 0; e < 8; ++e) {
                Vts[kcol + 2 * e][krow]     = (unsigned short)(w[e] & 0xFFFFu);
                Vts[kcol + 2 * e + 1][krow] = (unsigned short)(w[e] >> 16);
            }
        }
        if (j0 + 32 < S) stage_kv(j0 + 32);  // prefetch next chunk
        __syncthreads();  // Ks/Vts ready

        // Scores: 16x32 = two 16x16 WMMA accumulations over 4 d-chunks.
        v8f sc0 = zero8(), sc1 = zero8();
#pragma unroll
        for (int dc = 0; dc < 4; ++dc) {
            BFrag bk0, bk1;
            load_b_frag(&Ks[0][0], 128, 0, dc * 32, bk0);
            load_b_frag(&Ks[0][0], 128, 16, dc * 32, bk1);
            sc0 = wmma_bf16(qf[dc], bk0, sc0);
            sc1 = wmma_bf16(qf[dc], bk1, sc1);
        }

        // Online softmax. Each row lives in one VGPR across a 16-lane half.
        const int half = lane >> 4;
        const int nl   = lane & 15;
#pragma unroll
        for (int i = 0; i < 8; ++i) {
            float s0 = sc0[i] * scale;
            float s1 = sc1[i] * scale;
            float mx = fmaxf(s0, s1);
#pragma unroll
            for (int msk = 1; msk <= 8; msk <<= 1)
                mx = fmaxf(mx, __shfl_xor(mx, msk, 32));
            const float mnew  = fmaxf(mrow[i], mx);
            const float alpha = __expf(mrow[i] - mnew);
            const float p0 = __expf(s0 - mnew);
            const float p1 = __expf(s1 - mnew);
            float ps = p0 + p1;
#pragma unroll
            for (int msk = 1; msk <= 8; msk <<= 1)
                ps += __shfl_xor(ps, msk, 32);
            srow[i] = srow[i] * alpha + ps;
            mrow[i] = mnew;
#pragma unroll
            for (int nt = 0; nt < 8; ++nt) oacc[nt][i] *= alpha;
            const int r = i + half * 8;
            Ps[wave][r][nl]      = (bf16_t)p0;
            Ps[wave][r][16 + nl] = (bf16_t)p1;
        }
        __syncthreads();  // P tiles visible

        // O += P(16x32) * V(32x128): 8 N-tiles of 16.
        BFrag ap;
        load_a_frag(&Ps[wave][0][0], 32, 0, 0, ap);
#pragma unroll
        for (int nt = 0; nt < 8; ++nt) {
            BFrag bv;
            load_b_frag((const bf16_t*)&Vts[0][0], 32, nt * 16, 0, bv);
            oacc[nt] = wmma_bf16(ap, bv, oacc[nt]);
        }
    }

    // Normalize and write attention output (bf16) in token-major [B*S][32*128] layout.
    const int half = lane >> 4;
    const int nl   = lane & 15;
#pragma unroll
    for (int i = 0; i < 8; ++i) {
        const float inv = 1.0f / srow[i];
        bf16_t* orow =
            o + (size_t)(b * S + sRow + i + half * 8) * HQ + h * 128;
#pragma unroll
        for (int nt = 0; nt < 8; ++nt)
            orow[nt * 16 + nl] = (bf16_t)(oacc[nt][i] * inv);
    }
}

// ---------------------------------------------------------------------------
// Host-side launch
// ---------------------------------------------------------------------------
extern "C" void kernel_launch(void* const* d_in, const int* in_sizes, int n_in,
                              void* d_out, int out_size, void* d_ws,
                              size_t ws_size, hipStream_t stream) {
    const float* x  = (const float*)d_in[0];
    const float* wq = (const float*)d_in[1];
    const float* wk = (const float*)d_in[2];
    const float* wv = (const float*)d_in[3];
    const float* wo = (const float*)d_in[4];

    const int H   = 4096;
    const int HKV = 1024;
    const int B   = 2;
    const long T  = in_sizes[0] / H;       // total tokens (B*S) = 4096
    const int S   = (int)(T / B);          // 2048

    // Workspace layout (bf16 buffers), 256B-aligned offsets.
    char* ws = (char*)d_ws;
    size_t off = 0;
    auto take = [&](size_t bytes) {
        char* p = ws + off;
        off += (bytes + 255) & ~(size_t)255;
        return p;
    };
    bf16_t* xb  = (bf16_t*)take((size_t)T * H * 2);
    bf16_t* wqb = (bf16_t*)take((size_t)H * H * 2);
    bf16_t* wkb = (bf16_t*)take((size_t)HKV * H * 2);
    bf16_t* wvb = (bf16_t*)take((size_t)HKV * H * 2);
    bf16_t* wob = (bf16_t*)take((size_t)H * H * 2);
    bf16_t* qb  = (bf16_t*)take((size_t)T * H * 2);
    bf16_t* kb  = (bf16_t*)take((size_t)T * HKV * 2);
    bf16_t* vb  = (bf16_t*)take((size_t)T * HKV * 2);
    bf16_t* ob  = (bf16_t*)take((size_t)T * H * 2);
    (void)ws_size;

    // 1) Convert inputs to bf16.
    f32_to_bf16_kernel<<<2048, 256, 0, stream>>>(x,  xb,  (size_t)T * H);
    f32_to_bf16_kernel<<<2048, 256, 0, stream>>>(wq, wqb, (size_t)H * H);
    f32_to_bf16_kernel<<<1024, 256, 0, stream>>>(wk, wkb, (size_t)HKV * H);
    f32_to_bf16_kernel<<<1024, 256, 0, stream>>>(wv, wvb, (size_t)HKV * H);
    f32_to_bf16_kernel<<<2048, 256, 0, stream>>>(wo, wob, (size_t)H * H);

    // 2) QKV projections: y = x @ W^T.
    dim3 gq(H / 128, (unsigned)(T / 128));
    gemm_bt_kernel<true><<<gq, 256, 0, stream>>>(xb, wqb, qb, (int)T, H, H);
    dim3 gk(HKV / 128, (unsigned)(T / 128));
    gemm_bt_kernel<true><<<gk, 256, 0, stream>>>(xb, wkb, kb, (int)T, HKV, H);
    gemm_bt_kernel<true><<<gk, 256, 0, stream>>>(xb, wvb, vb, (int)T, HKV, H);

    // 3) Flash attention.
    dim3 ga(S / 128, 32, B);
    attn_kernel<<<ga, 256, 0, stream>>>(qb, kb, vb, ob, S);

    // 4) Output projection -> f32 d_out.
    dim3 go(H / 128, (unsigned)(T / 128));
    gemm_bt_kernel<false><<<go, 256, 0, stream>>>(ob, wob, (float*)d_out,
                                                  (int)T, H, H);
}
